// MDTA_52398601011972
// MI455X (gfx1250) — matrix operand order repaired and checked by reading, not compiled, not run on hardware
//
#include <hip/hip_runtime.h>

typedef __attribute__((ext_vector_type(16))) __bf16 bf16x16;
typedef __attribute__((ext_vector_type(8)))  __bf16 bf16x8;
typedef __attribute__((ext_vector_type(8)))  float  f32x8;
typedef __attribute__((ext_vector_type(4)))  unsigned int u32x4;
typedef __attribute__((ext_vector_type(4)))  int i32x4;
typedef __attribute__((ext_vector_type(8)))  int i32x8;

constexpr int C_DIM   = 128;
constexpr int N_SEQ   = 4096;
constexpr int N_BATCH = 4;

__device__ __forceinline__ unsigned short f2bf_u(float f) {
    union { float f; unsigned int u; } v; v.f = f;
    unsigned int u = v.u;
    return (unsigned short)((u + 0x7fffu + ((u >> 16) & 1u)) >> 16);
}
__device__ __forceinline__ __bf16 f2bf(float f) {
    union { unsigned short u; __bf16 b; } v; v.u = f2bf_u(f);
    return v.b;
}
// Build a v16bf fragment from two contiguous 8-element (16B) chunks.
__device__ __forceinline__ bf16x16 frag_from(const unsigned short* p0,
                                             const unsigned short* p1) {
    bf16x8 lo = *(const bf16x8*)p0;
    bf16x8 hi = *(const bf16x8*)p1;
    bf16x16 r;
#pragma unroll
    for (int i = 0; i < 8; i++) { r[i] = lo[i]; r[i + 8] = hi[i]; }
    return r;
}

// ---------------------------------------------------------------------------
// TDM: issue a 2-D tensor_load_to_lds (bf16 elements). Descriptor fields per
// CDNA5 ISA. Groups 2/3 zero (2-D tile).
// ---------------------------------------------------------------------------
__device__ __forceinline__ void tdm_load_2d(unsigned lds_off,
                                            const unsigned short* gptr,
                                            unsigned tensor_d0, unsigned tensor_d1,
                                            unsigned tile_d0,   unsigned tile_d1,
                                            unsigned stride0) {
    unsigned long long ga = (unsigned long long)(size_t)gptr;
    u32x4 g0;
    g0[0] = 1u;                                           // count=1, is_restore=0
    g0[1] = lds_off;                                      // lds_addr (bytes)
    g0[2] = (unsigned)(ga & 0xffffffffu);                 // global_addr[31:0]
    g0[3] = (unsigned)((ga >> 32) & 0x1ffffffu) | (2u << 30);  // addr[56:32] | type=2
    i32x8 g1;
    g1[0] = (int)(1u << 16);                              // data_size=1 (2 bytes)
    g1[1] = (int)((tensor_d0 & 0xffffu) << 16);           // tensor_dim0[15:0]
    g1[2] = (int)((tensor_d0 >> 16) | ((tensor_d1 & 0xffffu) << 16));
    g1[3] = (int)((tensor_d1 >> 16) | (tile_d0 << 16));   // tile_dim0
    g1[4] = (int)tile_d1;                                 // tile_dim1 (tile_dim2=0)
    g1[5] = (int)stride0;                                 // tensor_dim0_stride[31:0]
    g1[6] = 0;
    g1[7] = 0;
    i32x4 z4 = {0, 0, 0, 0};
#if defined(__clang_major__) && (__clang_major__ >= 23)
    i32x8 z8 = {0, 0, 0, 0, 0, 0, 0, 0};
    __builtin_amdgcn_tensor_load_to_lds(g0, g1, z4, z4, z8, 0);
#else
    __builtin_amdgcn_tensor_load_to_lds(g0, g1, z4, z4, 0);
#endif
}

__device__ __forceinline__ unsigned lds_off_of(const void* p) {
    // generic shared pointer: low 32 bits are the LDS byte offset
    return (unsigned)(size_t)p;
}

// ---------------------------------------------------------------------------
// Kernel 0: transpose + bf16-convert weights.
// ---------------------------------------------------------------------------
__global__ void prep_weights(const float* __restrict__ Wfc,
                             const float* __restrict__ Wout,
                             unsigned short* __restrict__ WtFc,
                             unsigned short* __restrict__ WtOut) {
    int id = blockIdx.x * blockDim.x + threadIdx.x;
    int n1 = 3 * C_DIM * C_DIM;
    if (id < n1) {
        int j = id / C_DIM, i = id % C_DIM;
        WtFc[j * C_DIM + i] = f2bf_u(Wfc[i * (3 * C_DIM) + j]);
    } else if (id < n1 + C_DIM * C_DIM) {
        int t = id - n1;
        int j = t / C_DIM, i = t % C_DIM;
        WtOut[j * C_DIM + i] = f2bf_u(Wout[i * C_DIM + j]);
    }
}

// ---------------------------------------------------------------------------
// Kernel 1: qkv = x @ W_fc + b_fc.  Q,K row-major bf16; V stored transposed.
// ---------------------------------------------------------------------------
__global__ __launch_bounds__(128) void qkv_proj(
        const float* __restrict__ x, const unsigned short* __restrict__ WtFc,
        const float* __restrict__ bfc,
        unsigned short* __restrict__ Q, unsigned short* __restrict__ K,
        unsigned short* __restrict__ Vt) {
    const int wave = threadIdx.x >> 5, lane = threadIdx.x & 31;
    const int r = lane & 15, hi = lane >> 4;
    const int rowbase = blockIdx.x * 64 + wave * 16;

    bf16x16 af[4];
#pragma unroll
    for (int kc = 0; kc < 4; kc++) {
        const float* p = x + (size_t)(rowbase + r) * C_DIM + kc * 32 + hi * 8;
        bf16x16 a;
#pragma unroll
        for (int i = 0; i < 8; i++) { a[i] = f2bf(p[i]); a[i + 8] = f2bf(p[i + 16]); }
        af[kc] = a;
    }

#pragma unroll
    for (int ct = 0; ct < 24; ct++) {
        f32x8 acc = {};
#pragma unroll
        for (int kc = 0; kc < 4; kc++) {
            const unsigned short* wp =
                WtFc + (size_t)(ct * 16 + r) * C_DIM + kc * 32 + hi * 8;
            bf16x16 bfrag = frag_from(wp, wp + 16);
            acc = __builtin_amdgcn_wmma_f32_16x16x32_bf16(
                false, af[kc], false, bfrag, (short)0, acc, false, false);
        }
        int col = ct * 16 + r;
        float bias = bfc[col];
#pragma unroll
        for (int j = 0; j < 8; j++) {
            int row = rowbase + j + 8 * hi;
            unsigned short bv = f2bf_u(acc[j] + bias);
            if (col < C_DIM) {
                Q[(size_t)row * C_DIM + col] = bv;
            } else if (col < 2 * C_DIM) {
                K[(size_t)row * C_DIM + (col - C_DIM)] = bv;
            } else {
                int b = row / N_SEQ, n = row % N_SEQ;
                Vt[((size_t)b * C_DIM + (col - 2 * C_DIM)) * N_SEQ + n] = bv;
            }
        }
    }
}

// ---------------------------------------------------------------------------
// Kernel 2: flash attention, 64-key blocks, TDM double-buffered K/V staging.
// 4 waves/block, each owns a 16-row query strip; all share the staged tiles.
// ---------------------------------------------------------------------------
__global__ __launch_bounds__(128) void flash_attn(
        const unsigned short* __restrict__ Q, const unsigned short* __restrict__ K,
        const unsigned short* __restrict__ Vt, const float* __restrict__ scale,
        unsigned short* __restrict__ Oattn) {
    __shared__ alignas(128) unsigned short ldsK[2][64 * 128];   // 2 x 16 KB
    __shared__ alignas(128) unsigned short ldsV[2][128 * 64];   // 2 x 16 KB
    __shared__ alignas(16)  unsigned short ldsP[4][16][64];     // 8 KB

    const int wave = threadIdx.x >> 5, lane = threadIdx.x & 31;
    const int r = lane & 15, hi = lane >> 4;
    const int QB = N_SEQ / 64;
    const int b = blockIdx.x / QB, qb = blockIdx.x % QB;
    const int qrow = qb * 64 + wave * 16;
    const float comb = 0.08838834764831845f / scale[0];  // 1/sqrt(128) / scale

    // Resident Q fragments: 16 rows x 128 k
    bf16x16 qf[4];
#pragma unroll
    for (int kc = 0; kc < 4; kc++) {
        const unsigned short* p =
            Q + ((size_t)(b * N_SEQ + qrow + r)) * C_DIM + kc * 32 + hi * 8;
        qf[kc] = frag_from(p, p + 16);
    }

    f32x8 o[8];
    const f32x8 zf = {};
#pragma unroll
    for (int t = 0; t < 8; t++) o[t] = zf;
    float m[8], l[8];
#pragma unroll
    for (int j = 0; j < 8; j++) { m[j] = -3.0e38f; l[j] = 0.0f; }

    const int NIT = N_SEQ / 64;
    // Prologue: stage key-block 0 into buffer 0 (wave 0 drives the TDM).
    if (wave == 0) {
        tdm_load_2d(lds_off_of(&ldsK[0][0]),
                    K + (size_t)(b * N_SEQ) * C_DIM,
                    C_DIM, N_BATCH * N_SEQ, C_DIM, 64, C_DIM);
        tdm_load_2d(lds_off_of(&ldsV[0][0]),
                    Vt + (size_t)(b * C_DIM) * N_SEQ,
                    N_SEQ, C_DIM, 64, C_DIM, N_SEQ);
    }

    for (int it = 0; it < NIT; ++it) {
        const int buf = it & 1;
        if (wave == 0) __builtin_amdgcn_s_wait_tensorcnt(0);
        __syncthreads();                      // staged tiles visible to all waves
        if (wave == 0 && (it + 1) < NIT) {    // prefetch next block into other buf
            int j1 = (it + 1) * 64;
            tdm_load_2d(lds_off_of(&ldsK[buf ^ 1][0]),
                        K + ((size_t)(b * N_SEQ + j1)) * C_DIM,
                        C_DIM, N_BATCH * N_SEQ, C_DIM, 64, C_DIM);
            tdm_load_2d(lds_off_of(&ldsV[buf ^ 1][0]),
                        Vt + (size_t)(b * C_DIM) * N_SEQ + j1,
                        N_SEQ, C_DIM, 64, C_DIM, N_SEQ);
        }

        // ---- S = Q K^T : four 16-col tiles from the staged K tile ----
        f32x8 st[4];
#pragma unroll
        for (int t = 0; t < 4; t++) {
            f32x8 s = {};
            const unsigned short* kp = &ldsK[buf][(t * 16 + r) * C_DIM + hi * 8];
#pragma unroll
            for (int kc = 0; kc < 4; kc++) {
                bf16x16 bfrag = frag_from(kp + kc * 32, kp + kc * 32 + 16);
                s = __builtin_amdgcn_wmma_f32_16x16x32_bf16(
                    false, qf[kc], false, bfrag, (short)0, s, false, false);
            }
            st[t] = s;
        }
        // ---- online softmax over 64 keys ----
#pragma unroll
        for (int j = 0; j < 8; j++) {
            float sv[4];
#pragma unroll
            for (int t = 0; t < 4; t++) sv[t] = st[t][j] * comb;
            float mx = fmaxf(fmaxf(sv[0], sv[1]), fmaxf(sv[2], sv[3]));
#pragma unroll
            for (int off = 8; off >= 1; off >>= 1)
                mx = fmaxf(mx, __shfl_xor(mx, off, 32));
            float mn = fmaxf(m[j], mx);
            float alpha = __expf(m[j] - mn);
            float rs = 0.0f;
#pragma unroll
            for (int t = 0; t < 4; t++) {
                float p = __expf(sv[t] - mn);
                rs += p;
                ldsP[wave][j + 8 * hi][t * 16 + r] = f2bf_u(p);
            }
#pragma unroll
            for (int off = 8; off >= 1; off >>= 1)
                rs += __shfl_xor(rs, off, 32);
            m[j] = mn;
            l[j] = l[j] * alpha + rs;
#pragma unroll
            for (int t = 0; t < 8; t++) o[t][j] *= alpha;
        }
        asm volatile("s_wait_dscnt 0" ::: "memory");  // P in LDS before re-read
        // ---- P back in A-fragment layout (16 x 64 -> two K=32 fragments) ----
        bf16x16 pf[2];
#pragma unroll
        for (int kc2 = 0; kc2 < 2; kc2++) {
            const unsigned short* pp = &ldsP[wave][r][kc2 * 32 + hi * 8];
            pf[kc2] = frag_from(pp, pp + 16);
        }
        // ---- O += P * V from the staged V^T tile ----
#pragma unroll
        for (int ct = 0; ct < 8; ct++) {
#pragma unroll
            for (int kc2 = 0; kc2 < 2; kc2++) {
                const unsigned short* vp =
                    &ldsV[buf][(ct * 16 + r) * 64 + kc2 * 32 + hi * 8];
                bf16x16 bfrag = frag_from(vp, vp + 16);
                o[ct] = __builtin_amdgcn_wmma_f32_16x16x32_bf16(
                    false, pf[kc2], false, bfrag, (short)0, o[ct], false, false);
            }
        }
        // All our LDS reads of this iteration must retire before we arrive at
        // the next top-of-loop barrier (wave0 then overwrites buf^1).
        asm volatile("s_wait_dscnt 0" ::: "memory");
    }

    // ---- normalize by l and write O_attn (bf16) ----
#pragma unroll
    for (int j = 0; j < 8; j++) {
        float rcp = 1.0f / l[j];
        size_t base = ((size_t)(b * N_SEQ + qrow + j + 8 * hi)) * C_DIM;
#pragma unroll
        for (int ct = 0; ct < 8; ct++) {
            Oattn[base + ct * 16 + r] = f2bf_u(o[ct][j] * rcp);
        }
    }
}

// ---------------------------------------------------------------------------
// Kernel 3: out = O_attn @ W_out + b_out  (fp32 output)
// ---------------------------------------------------------------------------
__global__ __launch_bounds__(128) void out_proj(
        const unsigned short* __restrict__ Oattn,
        const unsigned short* __restrict__ WtOut,
        const float* __restrict__ bout, float* __restrict__ out) {
    const int wave = threadIdx.x >> 5, lane = threadIdx.x & 31;
    const int r = lane & 15, hi = lane >> 4;
    const int rowbase = blockIdx.x * 64 + wave * 16;

    bf16x16 af[4];
#pragma unroll
    for (int kc = 0; kc < 4; kc++) {
        const unsigned short* p =
            Oattn + (size_t)(rowbase + r) * C_DIM + kc * 32 + hi * 8;
        af[kc] = frag_from(p, p + 16);
    }
#pragma unroll
    for (int ct = 0; ct < 8; ct++) {
        f32x8 acc = {};
#pragma unroll
        for (int kc = 0; kc < 4; kc++) {
            const unsigned short* wp =
                WtOut + (size_t)(ct * 16 + r) * C_DIM + kc * 32 + hi * 8;
            bf16x16 bfrag = frag_from(wp, wp + 16);
            acc = __builtin_amdgcn_wmma_f32_16x16x32_bf16(
                false, af[kc], false, bfrag, (short)0, acc, false, false);
        }
        int col = ct * 16 + r;
        float bias = bout[col];
#pragma unroll
        for (int j = 0; j < 8; j++) {
            out[(size_t)(rowbase + j + 8 * hi) * C_DIM + col] = acc[j] + bias;
        }
    }
}

// ---------------------------------------------------------------------------
extern "C" void kernel_launch(void* const* d_in, const int* in_sizes, int n_in,
                              void* d_out, int out_size, void* d_ws, size_t ws_size,
                              hipStream_t stream) {
    (void)in_sizes; (void)n_in; (void)out_size; (void)ws_size;
    const float* x     = (const float*)d_in[0];
    const float* Wfc   = (const float*)d_in[1];
    const float* bfc   = (const float*)d_in[2];
    const float* Wout  = (const float*)d_in[3];
    const float* bout  = (const float*)d_in[4];
    const float* scale = (const float*)d_in[5];

    char* ws = (char*)d_ws;
    unsigned short* WtFc  = (unsigned short*)ws;
    unsigned short* WtOut = (unsigned short*)(ws + 3 * C_DIM * C_DIM * 2);
    size_t off = (size_t)(3 * C_DIM * C_DIM + C_DIM * C_DIM) * 2;
    off = (off + 255) & ~(size_t)255;
    const size_t qkvElems = (size_t)N_BATCH * N_SEQ * C_DIM;
    unsigned short* Q  = (unsigned short*)(ws + off);
    unsigned short* K  = Q + qkvElems;
    unsigned short* Vt = K + qkvElems;
    unsigned short* Oa = Vt + qkvElems;

    int totW = 3 * C_DIM * C_DIM + C_DIM * C_DIM;
    prep_weights<<<(totW + 255) / 256, 256, 0, stream>>>(Wfc, Wout, WtFc, WtOut);

    int rowblocks = (N_BATCH * N_SEQ) / 64;
    qkv_proj<<<rowblocks, 128, 0, stream>>>(x, WtFc, bfc, Q, K, Vt);
    flash_attn<<<N_BATCH * (N_SEQ / 64), 128, 0, stream>>>(Q, K, Vt, scale, Oa);
    out_proj<<<rowblocks, 128, 0, stream>>>(Oa, WtOut, bout, (float*)d_out);
}
